// SSA_41609643163839
// MI455X (gfx1250) — compile-verified, matrix operand order checked by reading, not run on hardware
//
#include <hip/hip_runtime.h>

// ---------------------------------------------------------------------------
// Problem constants (match the reference)
// ---------------------------------------------------------------------------
#define B_    2
#define S_    4096
#define D_    2048
#define H_    16
#define DN_   128
#define DR_   64
#define DQK_  192
#define DV_   128
#define QL_   512
#define KVL_  512
#define L_    64
#define BS_   (B_ * S_)               // 8192 token rows
#define NCH_  (S_ / L_)               // 64 chunks
#define SCALE_ 0.07216878364870323f   // 192^-0.5

typedef __attribute__((ext_vector_type(16))) __bf16 v16bf;
typedef __attribute__((ext_vector_type(8)))  float  v8f;

// ---------------------------------------------------------------------------
// WMMA helpers (CDNA5: D = A(16x32 bf16) * B(32x16 bf16) + C(16x16 f32))
// ---------------------------------------------------------------------------
__device__ __forceinline__ v8f wmma_bf16(v16bf a, v16bf b, v8f c) {
  return __builtin_amdgcn_wmma_f32_16x16x32_bf16(
      /*neg_a=*/false, a, /*neg_b=*/false, b,
      /*c_mod=*/(short)0, c, /*reuse_a=*/false, /*reuse_b=*/false);
}

// Load one 16x32 bf16 fragment (A layout; B uses the same layout with N rows)
// from a row-major LDS tile with leading dimension `ld` (elements).
// Lanes 0-15 -> rows 0-15, K 0..7 / 16..23; lanes 16-31 -> K 8..15 / 24..31.
__device__ __forceinline__ v16bf load_frag(const __bf16* p, int ld, int lane) {
  const int m    = lane & 15;
  const int hsel = lane >> 4;
  const __bf16* r = p + m * ld + hsel * 8;
  v16bf f;
  ((float4*)&f)[0] = *(const float4*)(r);        // K chunk 0 (8 bf16)
  ((float4*)&f)[1] = *(const float4*)(r + 16);   // K chunk 1 (8 bf16)
  return f;
}

// Accumulator element (vgpr r, lane l) maps to: m = (l>>4)*8 + r, n = l&15.

// ---------------------------------------------------------------------------
// CDNA5 async global->LDS copy (ASYNCcnt-tracked, no VGPR roundtrip).
//   global_load_async_to_lds_b128 vdst(LDS addr), vaddr(i32 offset), saddr
// LDS offset = low 32 bits of the flat address of a __shared__ location.
// ---------------------------------------------------------------------------
__device__ __forceinline__ void async_ld_b128(__bf16* lds_ptr, const void* base,
                                              unsigned byte_off) {
  const unsigned lds_off = (unsigned)(size_t)lds_ptr;
  asm volatile("global_load_async_to_lds_b128 %0, %1, %2"
               :: "v"(lds_off), "v"(byte_off), "s"(base)
               : "memory");
}
#define WAIT_ASYNC_LE4() asm volatile("s_wait_asynccnt 0x4" ::: "memory")
#define WAIT_ASYNC_0()   asm volatile("s_wait_asynccnt 0x0" ::: "memory")

// ---------------------------------------------------------------------------
// fp32 -> bf16 conversion (grid-stride)
// ---------------------------------------------------------------------------
__global__ __launch_bounds__(256) void cvt_f32_bf16(const float* __restrict__ in,
                                                    __bf16* __restrict__ out,
                                                    long n) {
  long i = (long)blockIdx.x * blockDim.x + threadIdx.x;
  const long stride = (long)gridDim.x * blockDim.x;
  for (; i < n; i += stride) out[i] = (__bf16)in[i];
}

// ---------------------------------------------------------------------------
// Tiled WMMA GEMM:  C[M,N] = A[M,K] * W[N,K]^T   (A, W bf16, C f32 or bf16)
// Block tile 128x128, K step 32, 256 threads = 8 waves (4x2 wave grid,
// each wave owns a 32x64 subtile = 2x4 WMMA tiles).
// Double-buffered LDS staging with async global->LDS B128 copies:
// each thread issues exactly 4 async loads per stage (2 for A, 2 for W),
// uniformly across waves, so `s_wait_asynccnt 4` after prefetching stage k+1
// guarantees stage k has landed (async loads complete in order).
// OOB weight rows (ragged N) are clamped to a valid row: they only feed
// output columns >= N, which the epilogue never stores.
// ---------------------------------------------------------------------------
template <int OUT_BF16>
__global__ __launch_bounds__(256)
void gemm_wmma(const __bf16* __restrict__ A, const __bf16* __restrict__ W,
               void* __restrict__ Cout, int M, int N, int K) {
  __shared__ __bf16 As[2][128 * 32];
  __shared__ __bf16 Ws[2][128 * 32];

  const int tid   = threadIdx.x;
  const int lane  = tid & 31;
  const int wave  = tid >> 5;
  const int waveM = wave >> 1;       // 0..3  -> 32-row strip
  const int waveN = wave & 1;        // 0..1  -> 64-col strip
  const int bm    = blockIdx.x * 128;
  const int bn    = blockIdx.y * 128;

  v8f acc[2][4];
#pragma unroll
  for (int mi = 0; mi < 2; ++mi)
#pragma unroll
    for (int ni = 0; ni < 4; ++ni) acc[mi][ni] = (v8f){};

  // Per-thread chunk mapping for staging: tile = 128 rows x 32 bf16 (64 B/row
  // = 4 x 16B chunks), 512 chunks total, 2 per thread per operand.
  const int c0_row = tid >> 2;              // chunk j=0: rows 0..63
  const int c0_q   = tid & 3;
  const int c1_row = (tid + 256) >> 2;      // chunk j=1: rows 64..127
  const int c1_q   = c0_q;

  auto stage = [&](int buf, int k0) {
    // A tile (rows always in range; M is a multiple of 128)
    async_ld_b128(&As[buf][c0_row * 32 + c0_q * 8], A,
                  (unsigned)((((size_t)(bm + c0_row)) * K + k0 + c0_q * 8) * 2));
    async_ld_b128(&As[buf][c1_row * 32 + c1_q * 8], A,
                  (unsigned)((((size_t)(bm + c1_row)) * K + k0 + c1_q * 8) * 2));
    // W tile (clamp ragged rows; uniform issue count across all lanes)
    const int w0 = (bn + c0_row < N) ? (bn + c0_row) : (N - 1);
    const int w1 = (bn + c1_row < N) ? (bn + c1_row) : (N - 1);
    async_ld_b128(&Ws[buf][c0_row * 32 + c0_q * 8], W,
                  (unsigned)((((size_t)w0) * K + k0 + c0_q * 8) * 2));
    async_ld_b128(&Ws[buf][c1_row * 32 + c1_q * 8], W,
                  (unsigned)((((size_t)w1) * K + k0 + c1_q * 8) * 2));
  };

  const int KT = K >> 5;
  stage(0, 0);

  for (int kt = 0; kt < KT; ++kt) {
    const int cur = kt & 1;
    if (kt + 1 < KT) {
      stage(cur ^ 1, (kt + 1) << 5);  // prefetch next stage into other buffer
      WAIT_ASYNC_LE4();               // newest 4 may be in flight; stage kt done
    } else {
      WAIT_ASYNC_0();
    }
    __syncthreads();                  // all waves' stage-kt data visible

    v16bf afrag[2];
#pragma unroll
    for (int mi = 0; mi < 2; ++mi)
      afrag[mi] = load_frag(&As[cur][(waveM * 32 + mi * 16) * 32], 32, lane);
#pragma unroll
    for (int ni = 0; ni < 4; ++ni) {
      v16bf bfrag = load_frag(&Ws[cur][(waveN * 64 + ni * 16) * 32], 32, lane);
#pragma unroll
      for (int mi = 0; mi < 2; ++mi)
        acc[mi][ni] = wmma_bf16(afrag[mi], bfrag, acc[mi][ni]);
    }
    __syncthreads();                  // compute done before buffer is re-staged
  }

  // Epilogue
  const int mbase = (lane >> 4) * 8;
  const int ncol  = lane & 15;
#pragma unroll
  for (int mi = 0; mi < 2; ++mi)
#pragma unroll
    for (int ni = 0; ni < 4; ++ni) {
      const int col = bn + waveN * 64 + ni * 16 + ncol;
      if (col < N) {
#pragma unroll
        for (int r = 0; r < 8; ++r) {
          const int row = bm + waveM * 32 + mi * 16 + mbase + r;
          const float v = acc[mi][ni][r];
          if (OUT_BF16)
            ((__bf16*)Cout)[(size_t)row * N + col] = (__bf16)v;
          else
            ((float*)Cout)[(size_t)row * N + col] = v;
        }
      }
    }
}

// ---------------------------------------------------------------------------
// RMS-norm over `cols` columns (fp32 accumulate), bf16 in -> bf16 out.
// One block per row; input row stride may exceed cols (kv_lat inside kvp).
// ---------------------------------------------------------------------------
__global__ __launch_bounds__(256)
void rmsnorm_bf16(const __bf16* __restrict__ in, int in_stride,
                  const float* __restrict__ w, __bf16* __restrict__ out,
                  int cols) {
  const int row = blockIdx.x;
  const __bf16* r = in + (size_t)row * in_stride;
  __shared__ float red[256];
  float s = 0.f;
  for (int c = threadIdx.x; c < cols; c += 256) {
    float v = (float)r[c];
    s += v * v;
  }
  red[threadIdx.x] = s;
  __syncthreads();
  for (int off = 128; off > 0; off >>= 1) {
    if (threadIdx.x < off) red[threadIdx.x] += red[threadIdx.x + off];
    __syncthreads();
  }
  const float inv = rsqrtf(red[0] / (float)cols + 1e-6f);
  for (int c = threadIdx.x; c < cols; c += 256)
    out[(size_t)row * cols + c] = (__bf16)((float)r[c] * inv * w[c]);
}

// ---------------------------------------------------------------------------
// Fused RoPE + (elu+1) + head-major relayout.
//   qproj  : [BS, H*DQK]  (per head: 128 nope + 64 pe)
//   kvproj : [BS, H*(DN+DV)] (per head: 128 k_nope + 128 v)
//   kvp    : [BS, KVL+DR] (cols 512..575 = shared k_pe)
// Outputs: qf/kf [B,H,S,DQK], vf [B,H,S,DV], all bf16.
// ---------------------------------------------------------------------------
__global__ __launch_bounds__(256)
void build_qkv(const __bf16* __restrict__ qproj,
               const __bf16* __restrict__ kvproj,
               const __bf16* __restrict__ kvp,
               const float* __restrict__ cosb, const float* __restrict__ sinb,
               __bf16* __restrict__ qf, __bf16* __restrict__ kf,
               __bf16* __restrict__ vf) {
  const int row = blockIdx.x;          // 0..BS-1
  const int b = row / S_;
  const int s = row - b * S_;

  for (int i = threadIdx.x; i < H_ * DQK_; i += 256) {
    const int h = i / DQK_;
    const int d = i - h * DQK_;
    const size_t dst = (((size_t)b * H_ + h) * S_ + s) * DQK_ + d;

    float qv, kvv;
    if (d < DN_) {
      qv  = (float)qproj[(size_t)row * (H_ * DQK_) + h * DQK_ + d];
      kvv = (float)kvproj[(size_t)row * (H_ * (DN_ + DV_)) + h * (DN_ + DV_) + d];
    } else {
      const int dd = d - DN_;
      const int p  = dd >> 1;
      const float c  = cosb[(size_t)s * (DR_ / 2) + p];
      const float sn = sinb[(size_t)s * (DR_ / 2) + p];
      // q_pe (per head)
      {
        const float xr = (float)qproj[(size_t)row * (H_ * DQK_) + h * DQK_ + DN_ + (p << 1)];
        const float xi = (float)qproj[(size_t)row * (H_ * DQK_) + h * DQK_ + DN_ + (p << 1) + 1];
        qv = (dd & 1) ? (xr * sn + xi * c) : (xr * c - xi * sn);
      }
      // k_pe (shared across heads)
      {
        const float xr = (float)kvp[(size_t)row * (KVL_ + DR_) + KVL_ + (p << 1)];
        const float xi = (float)kvp[(size_t)row * (KVL_ + DR_) + KVL_ + (p << 1) + 1];
        kvv = (dd & 1) ? (xr * sn + xi * c) : (xr * c - xi * sn);
      }
    }
    // elu(x)+1
    qv  = (qv  > 0.f) ? (qv  + 1.f) : __expf(qv);
    kvv = (kvv > 0.f) ? (kvv + 1.f) : __expf(kvv);
    qf[dst] = (__bf16)qv;
    kf[dst] = (__bf16)kvv;
  }

  for (int i = threadIdx.x; i < H_ * DV_; i += 256) {
    const int h = i / DV_;
    const int d = i - h * DV_;
    vf[(((size_t)b * H_ + h) * S_ + s) * DV_ + d] =
        kvproj[(size_t)row * (H_ * (DN_ + DV_)) + h * (DN_ + DV_) + DN_ + d];
  }
}

// ---------------------------------------------------------------------------
// SSA: chunked linear attention. One block per (b,h); 256 threads = 8 waves.
// Keeps the DQK x DV fp32 running state in LDS and scans the 64 chunks
// sequentially, doing all matmuls with bf16 WMMA:
//   scores = tril(qc . kc^T)            [64x64]   K=192
//   out    = qc . state + scores . vc [64x128]  K=192 / K=64
//   state += kc^T . vc                  [192x128] K=64
// ---------------------------------------------------------------------------
__global__ __launch_bounds__(256)
void ssa_wmma(const __bf16* __restrict__ qf, const __bf16* __restrict__ kf,
              const __bf16* __restrict__ vf,
              const float* __restrict__ init_state,
              __bf16* __restrict__ attn /* [BS, H*DV] */) {
  __shared__ float  stS[DQK_ * DV_];      // running state, f32   (98304 B)
  __shared__ __bf16 stT[DV_ * DQK_];      // state^T, bf16        (49152 B)
  __shared__ __bf16 qs[L_ * DQK_];        // chunk q              (24576 B)
  __shared__ __bf16 ks[L_ * DQK_];        // chunk k              (24576 B)
  __shared__ __bf16 kT[DQK_ * L_];        // chunk k^T            (24576 B)
  __shared__ __bf16 vT[DV_ * L_];         // chunk v^T            (16384 B)
  __shared__ __bf16 ss[L_ * L_];          // masked scores, bf16  ( 8192 B)

  const int bh   = blockIdx.x;            // b*H + h
  const int b    = bh >> 4;
  const int h    = bh & 15;
  const int tid  = threadIdx.x;
  const int lane = tid & 31;
  const int wave = tid >> 5;

  // Init state from input (zeros in reference, but honor it)
  for (int i = tid; i < DQK_ * DV_; i += 256)
    stS[i] = init_state[(size_t)bh * DQK_ * DV_ + i];
  __syncthreads();

  const __bf16* qh = qf + (size_t)bh * S_ * DQK_;
  const __bf16* kh = kf + (size_t)bh * S_ * DQK_;
  const __bf16* vh = vf + (size_t)bh * S_ * DV_;

  const int mbase = (lane >> 4) * 8;
  const int ncol  = lane & 15;

  for (int c = 0; c < NCH_; ++c) {
    // ---- Phase 1: stage chunk tiles + convert state to bf16^T ----
    for (int i = tid; i < L_ * DQK_; i += 256) {
      const __bf16 qv = qh[(size_t)c * L_ * DQK_ + i];
      const __bf16 kv = kh[(size_t)c * L_ * DQK_ + i];
      qs[i] = qv;
      ks[i] = kv;
      const int l = i / DQK_;
      const int d = i - l * DQK_;
      kT[d * L_ + l] = kv;
    }
    for (int i = tid; i < L_ * DV_; i += 256) {
      const int l = i >> 7;
      const int d = i & 127;
      vT[d * L_ + l] = vh[(size_t)c * L_ * DV_ + i];
    }
    for (int i = tid; i < DV_ * DQK_; i += 256) {
      const int n = i / DQK_;
      const int k = i - n * DQK_;
      stT[i] = (__bf16)stS[k * DV_ + n];
    }
    __syncthreads();

    // ---- Phase 2: scores = tril(qc . kc^T), 16 tiles, 2 per wave ----
#pragma unroll
    for (int t = 0; t < 2; ++t) {
      const int idx = wave * 2 + t;
      const int mt  = idx >> 2;
      const int nt  = idx & 3;
      v8f sacc = (v8f){};
#pragma unroll
      for (int k0 = 0; k0 < DQK_; k0 += 32) {
        v16bf af = load_frag(&qs[(mt * 16) * DQK_ + k0], DQK_, lane);
        v16bf bf = load_frag(&ks[(nt * 16) * DQK_ + k0], DQK_, lane);
        sacc = wmma_bf16(af, bf, sacc);
      }
      const int mrow = mt * 16 + mbase;
      const int nc   = nt * 16 + ncol;
#pragma unroll
      for (int r = 0; r < 8; ++r) {
        const float v = (nc <= mrow + r) ? sacc[r] : 0.f;
        ss[(mrow + r) * L_ + nc] = (__bf16)v;
      }
    }
    __syncthreads();

    // ---- Phase 3: out = qc.state + ss.vc, wave -> 16x64 strip ----
    {
      const int m0 = (wave >> 1) * 16;
      const int n0 = (wave & 1) * 64;
      v8f oacc[4];
#pragma unroll
      for (int ni = 0; ni < 4; ++ni) oacc[ni] = (v8f){};
#pragma unroll
      for (int k0 = 0; k0 < DQK_; k0 += 32) {
        v16bf af = load_frag(&qs[m0 * DQK_ + k0], DQK_, lane);
#pragma unroll
        for (int ni = 0; ni < 4; ++ni) {
          v16bf bf = load_frag(&stT[(n0 + ni * 16) * DQK_ + k0], DQK_, lane);
          oacc[ni] = wmma_bf16(af, bf, oacc[ni]);
        }
      }
#pragma unroll
      for (int k0 = 0; k0 < L_; k0 += 32) {
        v16bf af = load_frag(&ss[m0 * L_ + k0], L_, lane);
#pragma unroll
        for (int ni = 0; ni < 4; ++ni) {
          v16bf bf = load_frag(&vT[(n0 + ni * 16) * L_ + k0], L_, lane);
          oacc[ni] = wmma_bf16(af, bf, oacc[ni]);
        }
      }
      const int sg = c * L_ + m0 + mbase;
#pragma unroll
      for (int ni = 0; ni < 4; ++ni) {
        const int col = h * DV_ + n0 + ni * 16 + ncol;
#pragma unroll
        for (int r = 0; r < 8; ++r) {
          attn[((size_t)b * S_ + sg + r) * (H_ * DV_) + col] =
              (__bf16)(oacc[ni][r] * SCALE_);
        }
      }
    }

    // ---- Phase 4: state += kc^T . vc, 96 tiles, 12 per wave ----
#pragma unroll
    for (int t = 0; t < 12; ++t) {
      const int idx = wave * 12 + t;
      const int mt  = idx >> 3;
      const int nt  = idx & 7;
      const int mrow = mt * 16 + mbase;
      const int nc   = nt * 16 + ncol;
      v8f uacc;
#pragma unroll
      for (int r = 0; r < 8; ++r) uacc[r] = stS[(mrow + r) * DV_ + nc];
#pragma unroll
      for (int k0 = 0; k0 < L_; k0 += 32) {
        v16bf af = load_frag(&kT[(mt * 16) * L_ + k0], L_, lane);
        v16bf bf = load_frag(&vT[(nt * 16) * L_ + k0], L_, lane);
        uacc = wmma_bf16(af, bf, uacc);
      }
#pragma unroll
      for (int r = 0; r < 8; ++r) stS[(mrow + r) * DV_ + nc] = uacc[r];
    }
    __syncthreads();  // protect stS / staged tiles before next chunk
  }
}

// ---------------------------------------------------------------------------
// Host-side orchestration
// ---------------------------------------------------------------------------
extern "C" void kernel_launch(void* const* d_in, const int* in_sizes, int n_in,
                              void* d_out, int out_size, void* d_ws,
                              size_t ws_size, hipStream_t stream) {
  (void)in_sizes; (void)n_in; (void)out_size; (void)ws_size;

  const float* x      = (const float*)d_in[0];
  /* d_in[1] = start_pos (0, unused) */
  const float* cosb   = (const float*)d_in[2];
  const float* sinb   = (const float*)d_in[3];
  const float* wq_a   = (const float*)d_in[4];
  const float* wq_b   = (const float*)d_in[5];
  const float* wkv_a  = (const float*)d_in[6];
  const float* wkv_b  = (const float*)d_in[7];
  const float* wo     = (const float*)d_in[8];
  const float* qnw    = (const float*)d_in[9];
  const float* kvnw   = (const float*)d_in[10];
  const float* inist  = (const float*)d_in[11];

  // Workspace partition (bf16 staging + intermediates)
  size_t off = 0;
  auto bump = [&](size_t bytes) -> char* {
    char* p = (char*)d_ws + off;
    off += (bytes + 255) & ~(size_t)255;
    return p;
  };
  __bf16* x_bf    = (__bf16*)bump((size_t)BS_ * D_ * 2);
  __bf16* wqa_bf  = (__bf16*)bump((size_t)QL_ * D_ * 2);
  __bf16* wqb_bf  = (__bf16*)bump((size_t)H_ * DQK_ * QL_ * 2);
  __bf16* wkva_bf = (__bf16*)bump((size_t)(KVL_ + DR_) * D_ * 2);
  __bf16* wkvb_bf = (__bf16*)bump((size_t)H_ * (DN_ + DV_) * KVL_ * 2);
  __bf16* wo_bf   = (__bf16*)bump((size_t)D_ * H_ * DV_ * 2);
  __bf16* qlat    = (__bf16*)bump((size_t)BS_ * QL_ * 2);
  __bf16* qlatn   = (__bf16*)bump((size_t)BS_ * QL_ * 2);
  __bf16* qproj   = (__bf16*)bump((size_t)BS_ * H_ * DQK_ * 2);
  __bf16* kvp     = (__bf16*)bump((size_t)BS_ * (KVL_ + DR_) * 2);
  __bf16* kvlatn  = (__bf16*)bump((size_t)BS_ * KVL_ * 2);
  __bf16* kvproj  = (__bf16*)bump((size_t)BS_ * H_ * (DN_ + DV_) * 2);
  __bf16* qfeat   = (__bf16*)bump((size_t)B_ * H_ * S_ * DQK_ * 2);
  __bf16* kfeat   = (__bf16*)bump((size_t)B_ * H_ * S_ * DQK_ * 2);
  __bf16* vfeat   = (__bf16*)bump((size_t)B_ * H_ * S_ * DV_ * 2);
  __bf16* attn    = (__bf16*)bump((size_t)BS_ * H_ * DV_ * 2);

  // 1) fp32 -> bf16 staging
  cvt_f32_bf16<<<4096, 256, 0, stream>>>(x,     x_bf,    (long)BS_ * D_);
  cvt_f32_bf16<<<1024, 256, 0, stream>>>(wq_a,  wqa_bf,  (long)QL_ * D_);
  cvt_f32_bf16<<<1024, 256, 0, stream>>>(wq_b,  wqb_bf,  (long)H_ * DQK_ * QL_);
  cvt_f32_bf16<<<1024, 256, 0, stream>>>(wkv_a, wkva_bf, (long)(KVL_ + DR_) * D_);
  cvt_f32_bf16<<<1024, 256, 0, stream>>>(wkv_b, wkvb_bf, (long)H_ * (DN_ + DV_) * KVL_);
  cvt_f32_bf16<<<2048, 256, 0, stream>>>(wo,    wo_bf,   (long)D_ * H_ * DV_);

  // 2) q path: x @ wq_a^T -> rmsnorm -> @ wq_b^T
  gemm_wmma<1><<<dim3(BS_ / 128, QL_ / 128), 256, 0, stream>>>(
      x_bf, wqa_bf, qlat, BS_, QL_, D_);
  rmsnorm_bf16<<<BS_, 256, 0, stream>>>(qlat, QL_, qnw, qlatn, QL_);
  gemm_wmma<1><<<dim3(BS_ / 128, (H_ * DQK_) / 128), 256, 0, stream>>>(
      qlatn, wqb_bf, qproj, BS_, H_ * DQK_, QL_);

  // 3) kv path: x @ wkv_a^T (N=576, ragged) -> rmsnorm(first 512) -> @ wkv_b^T
  gemm_wmma<1><<<dim3(BS_ / 128, (KVL_ + DR_ + 127) / 128), 256, 0, stream>>>(
      x_bf, wkva_bf, kvp, BS_, KVL_ + DR_, D_);
  rmsnorm_bf16<<<BS_, 256, 0, stream>>>(kvp, KVL_ + DR_, kvnw, kvlatn, KVL_);
  gemm_wmma<1><<<dim3(BS_ / 128, (H_ * (DN_ + DV_)) / 128), 256, 0, stream>>>(
      kvlatn, wkvb_bf, kvproj, BS_, H_ * (DN_ + DV_), KVL_);

  // 4) RoPE + elu+1 + head-major relayout
  build_qkv<<<BS_, 256, 0, stream>>>(qproj, kvproj, kvp, cosb, sinb,
                                     qfeat, kfeat, vfeat);

  // 5) Chunked linear attention with LDS-resident state
  ssa_wmma<<<B_ * H_, 256, 0, stream>>>(qfeat, kfeat, vfeat, inist, attn);

  // 6) Output projection (fp32 out)
  gemm_wmma<0><<<dim3(BS_ / 128, D_ / 128), 256, 0, stream>>>(
      attn, wo_bf, (float*)d_out, BS_, D_, H_ * DV_);
}
